// ChunkConformerBlock_11596411699735
// MI455X (gfx1250) — matrix operand order repaired and checked by reading, not compiled, not run on hardware
//
#include <hip/hip_runtime.h>
#include <hip/hip_bf16.h>
#include <cstdint>
#include <cstddef>

#ifndef __has_builtin
#define __has_builtin(x) 0
#endif

#if __has_builtin(__builtin_amdgcn_global_load_async_to_lds_b128)
#define HAVE_ASYNC_LDS 1
#else
#define HAVE_ASYNC_LDS 0
#endif

typedef __attribute__((ext_vector_type(16))) __bf16 v16bf;
typedef __attribute__((ext_vector_type(8)))  float  v8f;

// Types matching the async-to-LDS builtin signature (from compiler diagnostic:
// param 0 is AS1 pointer to a 16-byte int vector; AS3 for the LDS side).
typedef int vs4i __attribute__((vector_size(4 * sizeof(int))));
typedef __attribute__((address_space(1))) vs4i* gv4i_ptr;
typedef __attribute__((address_space(3))) vs4i* lv4i_ptr;

namespace {

constexpr int kB = 16, kL = 1024, kD = 512, kH = 4, kDH = 128, kKc = 32;
constexpr int kM = kB * kL;  // 16384 token rows
constexpr float kEps = 1e-3f;

__device__ __forceinline__ float fast_sigmoid(float x) {
  return __builtin_amdgcn_rcpf(1.f + __expf(-x));
}
__device__ __forceinline__ float fast_silu(float x) {
  return x * __builtin_amdgcn_rcpf(1.f + __expf(-x));
}

// 16-byte global->LDS copy: async DMA path on CDNA5, VGPR-staged fallback.
__device__ __forceinline__ void copy16_g2l(const __hip_bfloat16* g, __hip_bfloat16* l) {
#if HAVE_ASYNC_LDS
  __builtin_amdgcn_global_load_async_to_lds_b128((gv4i_ptr)(g), (lv4i_ptr)(l), 0, 0);
#else
  *reinterpret_cast<uint4*>(l) = *reinterpret_cast<const uint4*>(g);
#endif
}

__device__ __forceinline__ void wait_async_zero() {
#if HAVE_ASYNC_LDS
#if __has_builtin(__builtin_amdgcn_s_wait_asynccnt)
  __builtin_amdgcn_s_wait_asynccnt(0);
#else
  asm volatile("s_wait_asynccnt 0x0" ::: "memory");
#endif
#endif
}

// ---------------------------------------------------------------------------
// One-time weight prep: Wt[n][k] = bf16(W[k][n])  (W fp32 [Kd, N])
// ---------------------------------------------------------------------------
__global__ __launch_bounds__(256) void wt_convert_kernel(
    const float* __restrict__ W, __hip_bfloat16* __restrict__ Wt, int Kd, int N) {
  const size_t idx = (size_t)blockIdx.x * 256 + threadIdx.x;
  const int k = (int)(idx / N), n = (int)(idx % N);
  Wt[(size_t)n * Kd + k] = __float2bfloat16(W[idx]);
}

// ---------------------------------------------------------------------------
// LayerNorm: one block (256 threads) per row of 512.
// ---------------------------------------------------------------------------
template <bool BF16OUT>
__global__ __launch_bounds__(256) void layernorm_kernel(
    const float* __restrict__ x, const float* __restrict__ g,
    const float* __restrict__ b, float* __restrict__ outf,
    __hip_bfloat16* __restrict__ outb) {
  __shared__ float ss[256], ss2[256];
  const int r = blockIdx.x, t = threadIdx.x;
  const float* xr = x + (size_t)r * kD;
  const float v0 = xr[t], v1 = xr[t + 256];
  ss[t] = v0 + v1;
  ss2[t] = v0 * v0 + v1 * v1;
  __syncthreads();
  for (int o = 128; o > 0; o >>= 1) {
    if (t < o) { ss[t] += ss[t + o]; ss2[t] += ss2[t + o]; }
    __syncthreads();
  }
  const float mean = ss[0] * (1.f / kD);
  const float var  = ss2[0] * (1.f / kD) - mean * mean;
  const float inv  = rsqrtf(var + kEps);
  const float y0 = (v0 - mean) * inv * g[t] + b[t];
  const float y1 = (v1 - mean) * inv * g[t + 256] + b[t + 256];
  if (BF16OUT) {
    outb[(size_t)r * kD + t]       = __float2bfloat16(y0);
    outb[(size_t)r * kD + t + 256] = __float2bfloat16(y1);
  } else {
    outf[(size_t)r * kD + t]       = y0;
    outf[(size_t)r * kD + t + 256] = y1;
  }
}

// ---------------------------------------------------------------------------
// WMMA bf16 GEMM: out = epilogue(A[M,Kd] @ Wt^T + bias), Wt bf16 [N, Kd].
// Block tile 128x128, BK=32, 256 threads = 8 waves, wave tile 32x64
// (2x4 grid of v_wmma_f32_16x16x32_bf16). Double-buffered LDS fed by
// async global->LDS DMA (ASYNCcnt) when available.
// MODE 0: fp32 out               MODE 1: SiLU -> bf16 out
// MODE 3: res + scale*val -> f32 MODE 4: BN affine + SiLU -> bf16 out
// ---------------------------------------------------------------------------
template <int MODE>
__global__ __launch_bounds__(256) void gemm_bf16_wmma(
    const __hip_bfloat16* __restrict__ A, const __hip_bfloat16* __restrict__ Wt,
    const float* __restrict__ bias, const float* __restrict__ res, float scale,
    float* __restrict__ outf, __hip_bfloat16* __restrict__ outb,
    const float* __restrict__ bng, const float* __restrict__ bnb,
    const float* __restrict__ bnm, const float* __restrict__ bnv,
    int M, int N, int Kd) {
  __shared__ __align__(16) __hip_bfloat16 As[2][128][32];
  __shared__ __align__(16) __hip_bfloat16 Bs[2][128][32];  // Bs[n][k]

  const int tid  = threadIdx.x;
  const int lane = tid & 31, wid = tid >> 5;
  const int wm = wid & 3, wn = wid >> 2;     // 4 row-waves x 2 col-waves
  const int m0 = blockIdx.y * 128, n0 = blockIdx.x * 128;
  const int row = lane & 15, kh = lane >> 4;

  // Per-thread 16B chunks: 512 chunks per tile / 256 threads = 2 each.
  const int r0 = tid >> 2,          kc0 = (tid & 3) << 3;
  const int r1 = (tid + 256) >> 2,  kc1 = (tid & 3) << 3;  // (c&3) identical

  const v8f vzero = {0.f, 0.f, 0.f, 0.f, 0.f, 0.f, 0.f, 0.f};
  v8f acc[2][4];
#pragma unroll
  for (int i = 0; i < 2; ++i)
#pragma unroll
    for (int j = 0; j < 4; ++j) acc[i][j] = vzero;

  const int nT = Kd >> 5;

  // Prologue: tile 0 -> buffer 0.
  {
    const int k0 = 0;
    copy16_g2l(&A [(size_t)(m0 + r0) * Kd + k0 + kc0], &As[0][r0][kc0]);
    copy16_g2l(&Wt[(size_t)(n0 + r0) * Kd + k0 + kc0], &Bs[0][r0][kc0]);
    copy16_g2l(&A [(size_t)(m0 + r1) * Kd + k0 + kc1], &As[0][r1][kc1]);
    copy16_g2l(&Wt[(size_t)(n0 + r1) * Kd + k0 + kc1], &Bs[0][r1][kc1]);
  }
  wait_async_zero();
  __syncthreads();

  for (int t = 0; t < nT; ++t) {
    const int cur = t & 1;
    if (t + 1 < nT) {  // issue DMA for next tile into the free buffer
      const int k0 = (t + 1) << 5;
      const int nb = cur ^ 1;
      copy16_g2l(&A [(size_t)(m0 + r0) * Kd + k0 + kc0], &As[nb][r0][kc0]);
      copy16_g2l(&Wt[(size_t)(n0 + r0) * Kd + k0 + kc0], &Bs[nb][r0][kc0]);
      copy16_g2l(&A [(size_t)(m0 + r1) * Kd + k0 + kc1], &As[nb][r1][kc1]);
      copy16_g2l(&Wt[(size_t)(n0 + r1) * Kd + k0 + kc1], &Bs[nb][r1][kc1]);
    }
    if (t + 2 < nT) {  // L2 prefetch for the tile after next
      const int k2 = (t + 2) << 5;
      __builtin_prefetch(&A [(size_t)(m0 + r0) * Kd + k2 + kc0], 0, 1);
      __builtin_prefetch(&Wt[(size_t)(n0 + r0) * Kd + k2 + kc0], 0, 1);
    }

    // B fragments: lane = column, lanes>=16 take K=16..31 half.
    v16bf bfrag[4];
#pragma unroll
    for (int tn = 0; tn < 4; ++tn) {
      const __hip_bfloat16* p = &Bs[cur][wn * 64 + tn * 16 + row][kh * 16];
      reinterpret_cast<uint4*>(&bfrag[tn])[0] = reinterpret_cast<const uint4*>(p)[0];
      reinterpret_cast<uint4*>(&bfrag[tn])[1] = reinterpret_cast<const uint4*>(p)[1];
    }
#pragma unroll
    for (int tm = 0; tm < 2; ++tm) {
      // A fragment: lane = row; runs K=[kh*16, +16).
      v16bf afrag;
      const __hip_bfloat16* pa = &As[cur][wm * 32 + tm * 16 + row][kh * 16];
      reinterpret_cast<uint4*>(&afrag)[0] = *reinterpret_cast<const uint4*>(pa);
      reinterpret_cast<uint4*>(&afrag)[1] = *reinterpret_cast<const uint4*>(pa + 8);
#pragma unroll
      for (int tn = 0; tn < 4; ++tn)
        acc[tm][tn] = __builtin_amdgcn_wmma_f32_16x16x32_bf16(
            false, afrag, false, bfrag[tn], (short)0, acc[tm][tn], false, false);
    }

    wait_async_zero();   // our DMA writes into the next buffer have landed
    __syncthreads();     // everyone done computing `cur` + all DMA visible
  }

  // Epilogue. C/D layout: lane&15 = N, VGPR v -> M = v (+8 for lanes>=16).
  const int nl = lane & 15, mh = (lane >> 4) * 8;
#pragma unroll
  for (int tm = 0; tm < 2; ++tm) {
#pragma unroll
    for (int tn = 0; tn < 4; ++tn) {
      const int n = n0 + wn * 64 + tn * 16 + nl;
      const float bv = bias[n];
      float bnscale = 0.f, bnshift = 0.f;
      if (MODE == 4) {
        bnscale = rsqrtf(bnv[n] + kEps) * bng[n];
        bnshift = bnb[n] - bnm[n] * bnscale;
      }
#pragma unroll
      for (int v = 0; v < 8; ++v) {
        const int m = m0 + wm * 32 + tm * 16 + mh + v;
        const size_t idx = (size_t)m * N + n;
        float val = acc[tm][tn][v] + bv;
        if (MODE == 0) {
          outf[idx] = val;
        } else if (MODE == 1) {
          outb[idx] = __float2bfloat16(fast_silu(val));
        } else if (MODE == 3) {
          outf[idx] = res[idx] + scale * val;
        } else if (MODE == 4) {
          val = val * bnscale + bnshift;
          outb[idx] = __float2bfloat16(fast_silu(val));
        }
      }
    }
  }
}

// ---------------------------------------------------------------------------
// Banded attention: window j in [max(i-6,0), min(i+2,L-1)], <=9 keys.
// One wave per (b, i, head); each lane owns 4 of the 128 head channels.
// ---------------------------------------------------------------------------
__global__ __launch_bounds__(128) void attention_kernel(
    const float* __restrict__ Q, const float* __restrict__ K,
    const float* __restrict__ V, __hip_bfloat16* __restrict__ O) {
  const int b = blockIdx.y, i = blockIdx.x;
  const int h = threadIdx.x >> 5, lane = threadIdx.x & 31;
  const size_t base = ((size_t)(b * kL + i)) * kD + h * kDH + lane * 4;
  const float4 q = *reinterpret_cast<const float4*>(&Q[base]);

  int lo = i - 6; if (lo < 0) lo = 0;
  int hi = i + 2; if (hi > kL - 1) hi = kL - 1;
  const int cnt = hi - lo + 1;  // uniform per block

  float s[9];
#pragma unroll
  for (int j = 0; j < 9; ++j) {
    const bool ok = j < cnt;
    const int jj = ok ? (lo + j) : hi;
    const float4 kv = *reinterpret_cast<const float4*>(
        &K[((size_t)(b * kL + jj)) * kD + h * kDH + lane * 4]);
    float p = q.x * kv.x + q.y * kv.y + q.z * kv.z + q.w * kv.w;
#pragma unroll
    for (int o = 16; o > 0; o >>= 1) p += __shfl_xor(p, o, 32);
    s[j] = ok ? p * 0.08838834764831845f : -1e30f;
  }
  float mx = s[0];
#pragma unroll
  for (int j = 1; j < 9; ++j) mx = fmaxf(mx, s[j]);
  float sum = 0.f;
#pragma unroll
  for (int j = 0; j < 9; ++j) { s[j] = __expf(s[j] - mx); sum += s[j]; }
  const float inv = __builtin_amdgcn_rcpf(sum);

  float o0 = 0.f, o1 = 0.f, o2 = 0.f, o3 = 0.f;
#pragma unroll
  for (int j = 0; j < 9; ++j) {
    const int jj = (j < cnt) ? (lo + j) : hi;
    const float a = s[j] * inv;
    const float4 vv = *reinterpret_cast<const float4*>(
        &V[((size_t)(b * kL + jj)) * kD + h * kDH + lane * 4]);
    o0 += a * vv.x; o1 += a * vv.y; o2 += a * vv.z; o3 += a * vv.w;
  }
  O[base + 0] = __float2bfloat16(o0);
  O[base + 1] = __float2bfloat16(o1);
  O[base + 2] = __float2bfloat16(o2);
  O[base + 3] = __float2bfloat16(o3);
}

// ---------------------------------------------------------------------------
// GLU: G[r,c] = P[r,c] * sigmoid(P[r,512+c]),  P:[M,1024], G:[M,512]
// ---------------------------------------------------------------------------
__global__ __launch_bounds__(256) void glu_kernel(const float* __restrict__ P,
                                                  float* __restrict__ G) {
  const size_t idx = (size_t)blockIdx.x * 256 + threadIdx.x;
  const size_t r = idx >> 9, c = idx & 511;
  const float a = P[r * 1024 + c];
  const float g = P[r * 1024 + 512 + c];
  G[idx] = a * fast_sigmoid(g);
}

// ---------------------------------------------------------------------------
// Causal depthwise conv, K=32: out[b,l,c] = sum_k x[b,l-31+k,c]*w[k,c]
// ---------------------------------------------------------------------------
__global__ __launch_bounds__(256) void dwconv_kernel(
    const float* __restrict__ G, const float* __restrict__ w,
    __hip_bfloat16* __restrict__ out) {
  const int b = blockIdx.y, l = blockIdx.x;
  for (int cc = threadIdx.x; cc < kD; cc += 256) {
    float acc = 0.f;
#pragma unroll 8
    for (int k = 0; k < kKc; ++k) {
      const int ls = l - (kKc - 1) + k;
      if (ls >= 0) acc += G[((size_t)(b * kL + ls)) * kD + cc] * w[k * kD + cc];
    }
    out[((size_t)(b * kL + l)) * kD + cc] = __float2bfloat16(acc);
  }
}

}  // namespace

// ---------------------------------------------------------------------------
extern "C" void kernel_launch(void* const* d_in, const int* in_sizes, int n_in,
                              void* d_out, int out_size, void* d_ws, size_t ws_size,
                              hipStream_t stream) {
  (void)in_sizes; (void)n_in; (void)out_size; (void)ws_size;
  const float* x      = (const float*)d_in[0];
  const float* ff1_g  = (const float*)d_in[1];
  const float* ff1_b  = (const float*)d_in[2];
  const float* ff1_w1 = (const float*)d_in[3];
  const float* ff1_b1 = (const float*)d_in[4];
  const float* ff1_w2 = (const float*)d_in[5];
  const float* ff1_b2 = (const float*)d_in[6];
  const float* ff2_g  = (const float*)d_in[7];
  const float* ff2_b  = (const float*)d_in[8];
  const float* ff2_w1 = (const float*)d_in[9];
  const float* ff2_b1 = (const float*)d_in[10];
  const float* ff2_w2 = (const float*)d_in[11];
  const float* ff2_b2 = (const float*)d_in[12];
  const float* mh_g   = (const float*)d_in[13];
  const float* mh_b   = (const float*)d_in[14];
  const float* wq     = (const float*)d_in[15];
  const float* bq     = (const float*)d_in[16];
  const float* wk     = (const float*)d_in[17];
  const float* bk     = (const float*)d_in[18];
  const float* wv     = (const float*)d_in[19];
  const float* bv     = (const float*)d_in[20];
  const float* wo     = (const float*)d_in[21];
  const float* bo     = (const float*)d_in[22];
  const float* cv_g   = (const float*)d_in[23];
  const float* cv_b   = (const float*)d_in[24];
  const float* pw1_w  = (const float*)d_in[25];
  const float* pw1_b  = (const float*)d_in[26];
  const float* dw_w   = (const float*)d_in[27];
  const float* sep_w  = (const float*)d_in[28];
  const float* sep_b  = (const float*)d_in[29];
  const float* bn_g   = (const float*)d_in[30];
  const float* bn_b   = (const float*)d_in[31];
  const float* bn_m   = (const float*)d_in[32];
  const float* bn_v   = (const float*)d_in[33];
  const float* pw2_w  = (const float*)d_in[34];
  const float* pw2_b  = (const float*)d_in[35];
  const float* ln_g   = (const float*)d_in[36];
  const float* ln_b   = (const float*)d_in[37];

  // Workspace arena (lifetime-based reuse).
  char* ws = (char*)d_ws;
  __hip_bfloat16* Abf = (__hip_bfloat16*)(ws + (size_t)0);          // 16MB: LN/O/C bf16
  char*           big = ws + ((size_t)16 << 20);                    // 64MB: H bf16 or P f32
  __hip_bfloat16* Hbf = (__hip_bfloat16*)big;
  float*          P   = (float*)big;
  float* Qf = (float*)(ws + ((size_t)80  << 20));                   // 32MB (also GLU out G)
  float* Kf = (float*)(ws + ((size_t)112 << 20));                   // 32MB (also S bf16)
  float* Vf = (float*)(ws + ((size_t)144 << 20));                   // 32MB
  float* h1 = (float*)(ws + ((size_t)176 << 20));                   // 32MB (also h3)
  float* h2 = (float*)(ws + ((size_t)208 << 20));                   // 32MB (also h4)
  float* Gf = Qf;
  __hip_bfloat16* Sbf = (__hip_bfloat16*)Kf;
  __hip_bfloat16* Cbf = Abf;
  float* h3 = h1;
  float* h4 = h2;

  // bf16 pre-transposed weights arena (one-shot converts, ~14MB).
  size_t woff = (size_t)240 << 20;
  auto alloc_wt = [&](size_t elems) {
    __hip_bfloat16* p = (__hip_bfloat16*)(ws + woff);
    woff += elems * sizeof(__hip_bfloat16);
    return p;
  };
  __hip_bfloat16* t_ff1w1 = alloc_wt((size_t)512 * 2048);
  __hip_bfloat16* t_ff1w2 = alloc_wt((size_t)2048 * 512);
  __hip_bfloat16* t_ff2w1 = alloc_wt((size_t)512 * 2048);
  __hip_bfloat16* t_ff2w2 = alloc_wt((size_t)2048 * 512);
  __hip_bfloat16* t_wq    = alloc_wt((size_t)512 * 512);
  __hip_bfloat16* t_wk    = alloc_wt((size_t)512 * 512);
  __hip_bfloat16* t_wv    = alloc_wt((size_t)512 * 512);
  __hip_bfloat16* t_wo    = alloc_wt((size_t)512 * 512);
  __hip_bfloat16* t_pw1   = alloc_wt((size_t)512 * 1024);
  __hip_bfloat16* t_sep   = alloc_wt((size_t)512 * 1024);
  __hip_bfloat16* t_pw2   = alloc_wt((size_t)1024 * 512);

  auto conv_wt = [&](const float* W, __hip_bfloat16* Wt, int Kd, int N) {
    wt_convert_kernel<<<((size_t)Kd * N) / 256, 256, 0, stream>>>(W, Wt, Kd, N);
  };
  conv_wt(ff1_w1, t_ff1w1, 512, 2048);
  conv_wt(ff1_w2, t_ff1w2, 2048, 512);
  conv_wt(ff2_w1, t_ff2w1, 512, 2048);
  conv_wt(ff2_w2, t_ff2w2, 2048, 512);
  conv_wt(wq,     t_wq,    512, 512);
  conv_wt(wk,     t_wk,    512, 512);
  conv_wt(wv,     t_wv,    512, 512);
  conv_wt(wo,     t_wo,    512, 512);
  conv_wt(pw1_w,  t_pw1,   512, 1024);
  conv_wt(sep_w,  t_sep,   512, 1024);
  conv_wt(pw2_w,  t_pw2,   1024, 512);

  const dim3 gN4(4, kM / 128), gN8(8, kM / 128), gN16(16, kM / 128);
  const dim3 lnG(kM);

  // ---- FF module 1: h1 = x + 0.5*(silu(LN(x)@w1+b1)@w2+b2)
  layernorm_kernel<true><<<lnG, 256, 0, stream>>>(x, ff1_g, ff1_b, nullptr, Abf);
  gemm_bf16_wmma<1><<<gN16, 256, 0, stream>>>(Abf, t_ff1w1, ff1_b1, nullptr, 0.f,
      nullptr, Hbf, nullptr, nullptr, nullptr, nullptr, kM, 2048, 512);
  gemm_bf16_wmma<3><<<gN4, 256, 0, stream>>>(Hbf, t_ff1w2, ff1_b2, x, 0.5f,
      h1, nullptr, nullptr, nullptr, nullptr, nullptr, kM, 512, 2048);

  // ---- MHSA: h2 = h1 + (banded_attn(LN(h1)))@wo + bo
  layernorm_kernel<true><<<lnG, 256, 0, stream>>>(h1, mh_g, mh_b, nullptr, Abf);
  gemm_bf16_wmma<0><<<gN4, 256, 0, stream>>>(Abf, t_wq, bq, nullptr, 0.f,
      Qf, nullptr, nullptr, nullptr, nullptr, nullptr, kM, 512, 512);
  gemm_bf16_wmma<0><<<gN4, 256, 0, stream>>>(Abf, t_wk, bk, nullptr, 0.f,
      Kf, nullptr, nullptr, nullptr, nullptr, nullptr, kM, 512, 512);
  gemm_bf16_wmma<0><<<gN4, 256, 0, stream>>>(Abf, t_wv, bv, nullptr, 0.f,
      Vf, nullptr, nullptr, nullptr, nullptr, nullptr, kM, 512, 512);
  attention_kernel<<<dim3(kL, kB), 128, 0, stream>>>(Qf, Kf, Vf, Abf);  // O -> Abf
  gemm_bf16_wmma<3><<<gN4, 256, 0, stream>>>(Abf, t_wo, bo, h1, 1.0f,
      h2, nullptr, nullptr, nullptr, nullptr, nullptr, kM, 512, 512);

  // ---- Conv module: h3 = h2 + pw2(silu(BN(sep(dwconv(GLU(pw1(LN(h2))))))))
  layernorm_kernel<true><<<lnG, 256, 0, stream>>>(h2, cv_g, cv_b, nullptr, Abf);
  gemm_bf16_wmma<0><<<gN8, 256, 0, stream>>>(Abf, t_pw1, pw1_b, nullptr, 0.f,
      P, nullptr, nullptr, nullptr, nullptr, nullptr, kM, 1024, 512);
  glu_kernel<<<(kM * kD) / 256, 256, 0, stream>>>(P, Gf);
  dwconv_kernel<<<dim3(kL, kB), 256, 0, stream>>>(Gf, dw_w, Cbf);
  gemm_bf16_wmma<4><<<gN8, 256, 0, stream>>>(Cbf, t_sep, sep_b, nullptr, 0.f,
      nullptr, Sbf, bn_g, bn_b, bn_m, bn_v, kM, 1024, 512);
  gemm_bf16_wmma<3><<<gN4, 256, 0, stream>>>(Sbf, t_pw2, pw2_b, h2, 1.0f,
      h3, nullptr, nullptr, nullptr, nullptr, nullptr, kM, 512, 1024);

  // ---- FF module 2: h4 = h3 + 0.5*(silu(LN(h3)@w1+b1)@w2+b2)
  layernorm_kernel<true><<<lnG, 256, 0, stream>>>(h3, ff2_g, ff2_b, nullptr, Abf);
  gemm_bf16_wmma<1><<<gN16, 256, 0, stream>>>(Abf, t_ff2w1, ff2_b1, nullptr, 0.f,
      nullptr, Hbf, nullptr, nullptr, nullptr, nullptr, kM, 2048, 512);
  gemm_bf16_wmma<3><<<gN4, 256, 0, stream>>>(Hbf, t_ff2w2, ff2_b2, h3, 0.5f,
      h4, nullptr, nullptr, nullptr, nullptr, nullptr, kM, 512, 2048);

  // ---- Final LayerNorm -> d_out (fp32)
  layernorm_kernel<false><<<lnG, 256, 0, stream>>>(h4, ln_g, ln_b, (float*)d_out, nullptr);
}
